// MapNet_52776558133729
// MI455X (gfx1250) — compile-verified, matrix operand order checked by reading, not run on hardware
//
#include <hip/hip_runtime.h>
#include <hip/hip_bf16.h>

// ---------------------------------------------------------------------------
// MapNet fusion net on MI455X (gfx1250, wave32, WMMA, async-LDS DMA).
//
// Memory-bound workload (random feat-row gathers dominate), so:
//  * feat is mirrored in bf16 (written by the GN kernels) -> gather traffic
//    is 256B/row instead of 512B/row; WMMA consumes bf16 natively.
//  * all 128x128 weights are pre-converted once to bf16, pre-transposed
//    N-major (~2MB, L2-resident) -> GEMM staging is pure 16B copies done
//    with global_load_async_to_lds_b128 (ASYNCcnt, CDNA5 TDM-lite path).
//  * GEMM: 256 thr = 8 waves, 64 rows/block, wave owns a 16-col slab,
//    4 M-tiles x 4 K-slices = 16 v_wmma_f32_16x16x32_bf16 per wave.
//  * EDGE=false: accumulators bounce through LDS, coalesced b128 stores.
//    EDGE=true : branch-free global_atomic_add_f32 scatter.
// ---------------------------------------------------------------------------

typedef __bf16 bf16_t;
typedef __bf16  v16bf __attribute__((ext_vector_type(16)));
typedef float   v8f   __attribute__((ext_vector_type(8)));

#define NN 131072
#define CC 128
#define BB 4
#define SS 6
#define EE 131072
#define ELE 32768
#define WMAT (CC * CC)   // 16384 elements per 128x128 matrix

// CDNA5 async global->LDS DMA (tracked by ASYNCcnt).
__device__ __forceinline__ void async_ld_b128(unsigned lds_off, unsigned long long gaddr)
{
    asm volatile("global_load_async_to_lds_b128 %0, %1, off"
                 :: "v"(lds_off), "v"(gaddr) : "memory");
}
__device__ __forceinline__ void wait_async0()
{
    asm volatile("s_wait_asynccnt 0x0" ::: "memory");
}

// ---------------------------------------------------------------------------
// Tiled GEMM over bf16 activations/weights, f32 accumulate.
//   EDGE=false: Y[rbase+r] = Xb[rbase+r] @ W      (coalesced stores)
//   EDGE=true : Y[sidx[e]] += Xb[gidx[e]] @ W[s]  (atomic scatter-add)
// blockIdx.y = scale s (weight slab + index slab). Row counts divisible by 64.
// ---------------------------------------------------------------------------
template<bool EDGE>
__global__ __launch_bounds__(256)
void mapnet_wmma_gemm(const bf16_t* __restrict__ Xb,   // [N,128] bf16
                      const int*    __restrict__ gidx, // [scales, rows] (EDGE)
                      const int*    __restrict__ sidx, // [scales, rows] (EDGE)
                      const bf16_t* __restrict__ Wb,   // [scales,128,128] N-major bf16
                      float*        __restrict__ Y,    // [N,128] f32
                      int idx_stride)
{
    __shared__ __align__(16) unsigned char smem[16384 + 32768];
    bf16_t* sA = (bf16_t*)smem;             // 16 KB: 64 rows x 128 bf16
    bf16_t* sW = (bf16_t*)(smem + 16384);   // 32 KB: weight, N-major
    float*  sC = (float*)smem;              // 32 KB: reused after compute (node path)
    __shared__ int sOut[64];

    const int t = threadIdx.x;
    const int s = blockIdx.y;
    const int rbase = blockIdx.x * 64;
    const bf16_t* Wsel = Wb + (size_t)s * WMAT;

    // ---- async DMA: weight slab, 128 B per thread ----
    {
        unsigned long long src = (unsigned long long)(uintptr_t)Wsel + (unsigned)t * 128u;
        unsigned dst = (unsigned)(uintptr_t)sW + (unsigned)t * 128u;
        #pragma unroll
        for (int j = 0; j < 8; ++j) async_ld_b128(dst + j * 16, src + j * 16);
    }
    // ---- async DMA: 64 activation rows (gathered for EDGE), 64 B per thread ----
    {
        int r   = t >> 2;
        int off = (t & 3) * 64;
        int src_row = EDGE ? gidx[(size_t)s * idx_stride + rbase + r] : (rbase + r);
        unsigned long long src =
            (unsigned long long)(uintptr_t)(Xb + (size_t)src_row * 128) + (unsigned)off;
        unsigned dst = (unsigned)(uintptr_t)sA + (unsigned)(r * 256 + off);
        #pragma unroll
        for (int j = 0; j < 4; ++j) async_ld_b128(dst + j * 16, src + j * 16);
    }
    if (EDGE && t < 64)
        sOut[t] = sidx[(size_t)s * idx_stride + rbase + t];
    wait_async0();
    __syncthreads();

    // ---- WMMA: wave wv owns columns [wv*16, wv*16+16) ----
    const int wv   = t >> 5;
    const int lane = t & 31;
    const int nb   = wv * 16;
    const int mlo  = lane & 15;          // A-frag row / B-frag col (ISA layout)
    const int k0   = (lane >> 4) * 8;    // K sub-offset per ISA A/B layout
    const int n    = lane & 15;          // C/D col
    const int mhi  = (lane >> 4) << 3;   // C/D row bias (+8 for lanes 16..31)

    // B fragments are invariant across M-tiles: load once (4 K-slices).
    v16bf bfr[4];
    {
        const bf16_t* brow = sW + (size_t)(nb + mlo) * 128;
        #pragma unroll
        for (int kb = 0; kb < 4; ++kb) {
            #pragma unroll
            for (int j = 0; j < 8; ++j) {
                bfr[kb][j]     = brow[kb * 32 + k0 + j];
                bfr[kb][8 + j] = brow[kb * 32 + k0 + 16 + j];
            }
        }
    }

    v8f acc[4];
    #pragma unroll
    for (int mt = 0; mt < 4; ++mt) {
        const bf16_t* arow = sA + (size_t)(mt * 16 + mlo) * 128;
        v8f c = {};
        #pragma unroll
        for (int kb = 0; kb < 4; ++kb) {
            v16bf a;
            #pragma unroll
            for (int j = 0; j < 8; ++j) {
                a[j]     = arow[kb * 32 + k0 + j];
                a[8 + j] = arow[kb * 32 + k0 + 16 + j];
            }
            c = __builtin_amdgcn_wmma_f32_16x16x32_bf16(
                    false, a, false, bfr[kb], (short)0, c, false, false);
        }
        acc[mt] = c;
    }

    if (EDGE) {
        // Branch-free scatter-add (global_atomic_add_f32, agent scope).
        #pragma unroll
        for (int mt = 0; mt < 4; ++mt) {
            #pragma unroll
            for (int p = 0; p < 8; ++p) {
                int orow = sOut[mt * 16 + p + mhi];
                __hip_atomic_fetch_add(Y + (size_t)orow * 128 + nb + n, acc[mt][p],
                                       __ATOMIC_RELAXED, __HIP_MEMORY_SCOPE_AGENT);
            }
        }
    } else {
        // Bounce through LDS for fully coalesced b128 stores.
        __syncthreads();   // everyone done reading sA/sW
        #pragma unroll
        for (int mt = 0; mt < 4; ++mt)
            #pragma unroll
            for (int p = 0; p < 8; ++p)
                sC[(size_t)(mt * 16 + p + mhi) * 128 + nb + n] = acc[mt][p];
        __syncthreads();
        int r  = t >> 2;
        int cb = (t & 3) * 32;
        const float* srow = sC + (size_t)r * 128 + cb;
        float* drow = Y + (size_t)(rbase + r) * 128 + cb;
        #pragma unroll
        for (int j = 0; j < 8; ++j)
            *(float4*)(drow + j * 4) = *(const float4*)(srow + j * 4);
    }
}

// ---------------------------------------------------------------------------
// One-time weight conversion: f32 [nmats][k][n] -> bf16 [nmats][n][k]
// ---------------------------------------------------------------------------
__global__ __launch_bounds__(256)
void mapnet_convert_w(const float* __restrict__ in, bf16_t* __restrict__ out, int nmats)
{
    int i = blockIdx.x * 256 + threadIdx.x;
    if (i >= nmats * WMAT) return;
    int m = i / WMAT, w = i % WMAT;
    int k = w >> 7, nn = w & 127;
    out[(size_t)m * WMAT + nn * 128 + k] = (bf16_t)in[i];
}

// ---------------------------------------------------------------------------
// Input layer: Yb[n,c] = bf16(relu(x[n,0]*W0[0,c] + x[n,1]*W0[1,c] + b0[c]))
// ---------------------------------------------------------------------------
__global__ __launch_bounds__(256)
void mapnet_embed(const float* __restrict__ X2, const float* __restrict__ W0,
                  const float* __restrict__ b0, bf16_t* __restrict__ Yb, int nrows)
{
    int i = blockIdx.x * 256 + threadIdx.x;
    if (i >= nrows * 128) return;
    int r = i >> 7, c = i & 127;
    float o = X2[r * 2 + 0] * W0[c] + X2[r * 2 + 1] * W0[128 + c] + b0[c];
    Yb[i] = (bf16_t)(o > 0.f ? o : 0.f);
}

// ---------------------------------------------------------------------------
// GroupNorm helpers: one wave32 per 128-ch row, __shfl_xor reductions.
// ---------------------------------------------------------------------------
__device__ __forceinline__ void wave_meanvar(const float v[4], float& mu, float& inv)
{
    float s  = v[0] + v[1] + v[2] + v[3];
    float ss = v[0]*v[0] + v[1]*v[1] + v[2]*v[2] + v[3]*v[3];
    #pragma unroll
    for (int m = 16; m > 0; m >>= 1) {
        s  += __shfl_xor(s,  m, 32);
        ss += __shfl_xor(ss, m, 32);
    }
    mu = s * (1.f / 128.f);
    float var = ss * (1.f / 128.f) - mu * mu;
    inv = rsqrtf(var + 1e-5f);
}

// feat2bf = bf16(relu(GN(x)*g+b))       (mid-block, bf16-only output)
__global__ __launch_bounds__(256)
void mapnet_gn_mid(const float* __restrict__ X, const float* __restrict__ g,
                   const float* __restrict__ b, bf16_t* __restrict__ Yb, int nrows)
{
    int wv = threadIdx.x >> 5, lane = threadIdx.x & 31;
    int row = blockIdx.x * 8 + wv;
    if (row >= nrows) return;
    const float* xr = X + (size_t)row * 128;
    float v[4];
    #pragma unroll
    for (int j = 0; j < 4; ++j) v[j] = xr[lane + j * 32];
    float mu, inv;  wave_meanvar(v, mu, inv);
    bf16_t* yr = Yb + (size_t)row * 128;
    #pragma unroll
    for (int j = 0; j < 4; ++j) {
        int c = lane + j * 32;
        float o = (v[j] - mu) * inv * g[c] + b[c];
        yr[c] = (bf16_t)(o > 0.f ? o : 0.f);
    }
}

// feat = relu(GN(x)*g+b + res), written as f32 AND bf16 mirror
__global__ __launch_bounds__(256)
void mapnet_gn_res(const float* __restrict__ X, const float* __restrict__ g,
                   const float* __restrict__ b, const float* __restrict__ res,
                   float* __restrict__ Y, bf16_t* __restrict__ Yb, int nrows)
{
    int wv = threadIdx.x >> 5, lane = threadIdx.x & 31;
    int row = blockIdx.x * 8 + wv;
    if (row >= nrows) return;
    const float* xr = X + (size_t)row * 128;
    float v[4];
    #pragma unroll
    for (int j = 0; j < 4; ++j) v[j] = xr[lane + j * 32];
    float mu, inv;  wave_meanvar(v, mu, inv);
    float*  yr  = Y  + (size_t)row * 128;
    bf16_t* yrb = Yb + (size_t)row * 128;
    #pragma unroll
    for (int j = 0; j < 4; ++j) {
        int c = lane + j * 32;
        float o = (v[j] - mu) * inv * g[c] + b[c] + res[(size_t)row * 128 + c];
        o = o > 0.f ? o : 0.f;
        yr[c] = o;  yrb[c] = (bf16_t)o;
    }
}

// feat = relu(GN(g1)*ga1+bt1 + GN(g2)*ga2+bt2), f32 + bf16 mirror
__global__ __launch_bounds__(256)
void mapnet_gn2_fuse(const float* __restrict__ G1, const float* __restrict__ G2,
                     const float* __restrict__ ga1, const float* __restrict__ bt1,
                     const float* __restrict__ ga2, const float* __restrict__ bt2,
                     float* __restrict__ Y, bf16_t* __restrict__ Yb, int nrows)
{
    int wv = threadIdx.x >> 5, lane = threadIdx.x & 31;
    int row = blockIdx.x * 8 + wv;
    if (row >= nrows) return;
    const float* x1 = G1 + (size_t)row * 128;
    const float* x2 = G2 + (size_t)row * 128;
    float a[4], d[4];
    #pragma unroll
    for (int j = 0; j < 4; ++j) { a[j] = x1[lane + j*32]; d[j] = x2[lane + j*32]; }
    float mu1, inv1, mu2, inv2;
    wave_meanvar(a, mu1, inv1);
    wave_meanvar(d, mu2, inv2);
    float*  yr  = Y  + (size_t)row * 128;
    bf16_t* yrb = Yb + (size_t)row * 128;
    #pragma unroll
    for (int j = 0; j < 4; ++j) {
        int c = lane + j * 32;
        float o = (a[j] - mu1) * inv1 * ga1[c] + bt1[c]
                + (d[j] - mu2) * inv2 * ga2[c] + bt2[c];
        o = o > 0.f ? o : 0.f;
        yr[c] = o;  yrb[c] = (bf16_t)o;
    }
}

// ---------------------------------------------------------------------------
// Host orchestration.
// ws: featA | featB | work1 | work2 (f32, 64MB ea) | bf0 | bf1 (bf16, 32MB ea)
//     | wbf (66 x 128x128 bf16 ~ 2.1MB)
// ---------------------------------------------------------------------------
extern "C" void kernel_launch(void* const* d_in, const int* in_sizes, int n_in,
                              void* d_out, int out_size, void* d_ws, size_t ws_size,
                              hipStream_t stream)
{
    const float* ctrs   = (const float*)d_in[0];
    const float* feats  = (const float*)d_in[1];
    const int*   pre_u  = (const int*)d_in[2];
    const int*   pre_v  = (const int*)d_in[3];
    const int*   suc_u  = (const int*)d_in[4];
    const int*   suc_v  = (const int*)d_in[5];
    const int*   left_u = (const int*)d_in[6];
    const int*   left_v = (const int*)d_in[7];
    const int*   right_u= (const int*)d_in[8];
    const int*   right_v= (const int*)d_in[9];
    const float* ic_w0 = (const float*)d_in[10];
    const float* ic_b0 = (const float*)d_in[11];
    const float* ic_w1 = (const float*)d_in[12];
    const float* ic_g  = (const float*)d_in[13];
    const float* ic_bt = (const float*)d_in[14];
    const float* if_w0 = (const float*)d_in[15];
    const float* if_b0 = (const float*)d_in[16];
    const float* if_w1 = (const float*)d_in[17];
    const float* if_g  = (const float*)d_in[18];
    const float* if_bt = (const float*)d_in[19];
    const float* ctr_w   = (const float*)d_in[20];
    const float* pre_w   = (const float*)d_in[21];
    const float* suc_w   = (const float*)d_in[22];
    const float* left_w  = (const float*)d_in[23];
    const float* right_w = (const float*)d_in[24];
    const float* norm_g  = (const float*)d_in[25];
    const float* norm_b  = (const float*)d_in[26];
    const float* ctr2_w  = (const float*)d_in[27];
    const float* ctr2_g  = (const float*)d_in[28];
    const float* ctr2_b  = (const float*)d_in[29];
    (void)in_sizes; (void)n_in; (void)out_size; (void)ws_size;

    const size_t NC = (size_t)NN * CC;
    float*  featA = (float*)d_ws;
    float*  featB = featA + NC;
    float*  work1 = featB + NC;
    float*  work2 = work1 + NC;
    bf16_t* bf0   = (bf16_t*)(work2 + NC);
    bf16_t* bf1   = bf0 + NC;
    bf16_t* wbf   = bf1 + NC;

    // bf16/N-major weight slabs
    bf16_t* wb_ic    = wbf;
    bf16_t* wb_if    = wbf + (size_t)1  * WMAT;
    bf16_t* wb_ctr   = wbf + (size_t)2  * WMAT;   // 4 mats
    bf16_t* wb_ctr2  = wbf + (size_t)6  * WMAT;   // 4 mats
    bf16_t* wb_left  = wbf + (size_t)10 * WMAT;   // 4 mats
    bf16_t* wb_right = wbf + (size_t)14 * WMAT;   // 4 mats
    bf16_t* wb_pre   = wbf + (size_t)18 * WMAT;   // 24 mats
    bf16_t* wb_suc   = wbf + (size_t)42 * WMAT;   // 24 mats

    dim3 blk(256);
    dim3 gEmbed((NN * CC + 255) / 256);
    dim3 gNode(NN / 64, 1);
    dim3 gEdge(EE / 64, SS);
    dim3 gLR(ELE / 64, 1);
    dim3 gGN(NN / 8);
    auto gCvt = [](int nm) { return dim3((unsigned)(nm * WMAT / 256)); };

    // ---- one-time weight conversion (cheap, L2-resident afterwards) ----
    mapnet_convert_w<<<gCvt(1),      blk, 0, stream>>>(ic_w1,   wb_ic,    1);
    mapnet_convert_w<<<gCvt(1),      blk, 0, stream>>>(if_w1,   wb_if,    1);
    mapnet_convert_w<<<gCvt(BB),     blk, 0, stream>>>(ctr_w,   wb_ctr,   BB);
    mapnet_convert_w<<<gCvt(BB),     blk, 0, stream>>>(ctr2_w,  wb_ctr2,  BB);
    mapnet_convert_w<<<gCvt(BB),     blk, 0, stream>>>(left_w,  wb_left,  BB);
    mapnet_convert_w<<<gCvt(BB),     blk, 0, stream>>>(right_w, wb_right, BB);
    mapnet_convert_w<<<gCvt(BB*SS),  blk, 0, stream>>>(pre_w,   wb_pre,   BB*SS);
    mapnet_convert_w<<<gCvt(BB*SS),  blk, 0, stream>>>(suc_w,   wb_suc,   BB*SS);

    // ---- input encoders ----
    mapnet_embed<<<gEmbed, blk, 0, stream>>>(ctrs,  ic_w0, ic_b0, bf0, NN);
    mapnet_embed<<<gEmbed, blk, 0, stream>>>(feats, if_w0, if_b0, bf1, NN);
    mapnet_wmma_gemm<false><<<gNode, blk, 0, stream>>>(bf0, nullptr, nullptr,
                                                       wb_ic, featA, 0);
    mapnet_wmma_gemm<false><<<gNode, blk, 0, stream>>>(bf1, nullptr, nullptr,
                                                       wb_if, featB, 0);
    // feat = relu(GN(g1)+GN(g2)) -> featA (f32 residual) + bf0 (bf16 mirror)
    mapnet_gn2_fuse<<<gGN, blk, 0, stream>>>(featA, featB, ic_g, ic_bt, if_g, if_bt,
                                             featA, bf0, NN);

    float* cur   = featA;   // f32 residual
    float* other = featB;
    // bf0 = bf16 mirror of cur (rewritten in place each block)

    for (int i = 0; i < BB; ++i) {
        // temp = feat @ ctr_w[i]
        mapnet_wmma_gemm<false><<<gNode, blk, 0, stream>>>(
            bf0, nullptr, nullptr, wb_ctr + (size_t)i * WMAT, work1, 0);
        // multi-scale gather -> WMMA -> atomic scatter-add
        mapnet_wmma_gemm<true><<<gEdge, blk, 0, stream>>>(
            bf0, pre_v, pre_u, wb_pre + (size_t)i * SS * WMAT, work1, EE);
        mapnet_wmma_gemm<true><<<gEdge, blk, 0, stream>>>(
            bf0, suc_v, suc_u, wb_suc + (size_t)i * SS * WMAT, work1, EE);
        mapnet_wmma_gemm<true><<<gLR, blk, 0, stream>>>(
            bf0, left_v, left_u, wb_left + (size_t)i * WMAT, work1, 0);
        mapnet_wmma_gemm<true><<<gLR, blk, 0, stream>>>(
            bf0, right_v, right_u, wb_right + (size_t)i * WMAT, work1, 0);
        // feat2 (bf16) = relu(GN(temp))
        mapnet_gn_mid<<<gGN, blk, 0, stream>>>(work1, norm_g + (size_t)i * CC,
                                               norm_b + (size_t)i * CC, bf1, NN);
        // t2 = feat2 @ ctr2_w[i]
        mapnet_wmma_gemm<false><<<gNode, blk, 0, stream>>>(
            bf1, nullptr, nullptr, wb_ctr2 + (size_t)i * WMAT, work2, 0);
        // feat = relu(GN(t2) + identity); last block writes d_out
        float* nxt = (i == BB - 1) ? (float*)d_out : other;
        mapnet_gn_res<<<gGN, blk, 0, stream>>>(work2, ctr2_g + (size_t)i * CC,
                                               ctr2_b + (size_t)i * CC, cur,
                                               nxt, bf0, NN);
        if (i < BB - 1) { float* tp = cur; cur = nxt; other = tp; }
    }
}
